// SimpleRNN_20899310862489
// MI455X (gfx1250) — compile-verified, matrix operand order checked by reading, not run on hardware
//
#include <hip/hip_runtime.h>
#include <hip/hip_bf16.h>
#include <math.h>

// ---------------------------------------------------------------------------
// Fused 3-GRU RNN scan for MI455X (gfx1250, wave32, WMMA f16->f32).
//   sequence (512,2048,132) f32 -> hidden (512,2048,64) f32
// One wave owns 16 batch rows for all 512 timesteps. Weights pre-packed into
// WMMA B-fragment layout in LDS (f16, zero-padded). No barriers in main loop.
// Elementwise GRU math is branch-free using native v_rcp_f32 / v_tanh_f32.
// ---------------------------------------------------------------------------

typedef __attribute__((ext_vector_type(16))) _Float16 v16h;
typedef __attribute__((ext_vector_type(8)))  float    v8f;

#define T_STEPS 512
#define BATCH   2048
#define DIN     132
#define WAVES   4

// LDS layout (halves for weights/scratch, floats for biases)
#define OFF_WP1X 0        // [KS=2][NT=6]  p1 Wih^T   6144 h
#define OFF_WP1H 6144     // [KS=1][NT=6]  p1 Whh^T   3072 h
#define OFF_WP2X 9216     // 6144 h
#define OFF_WP2H 15360    // 3072 h
#define OFF_WG   18432    // [KS=2][NT=1]  g_W^T      1024 h
#define OFF_WRX  19456    // [KS=2][NT=12] r Wih^T   12288 h
#define OFF_WRH  31744    // [KS=2][NT=12] r Whh^T   12288 h
#define W_HALVES 44032
#define BIAS_BYTE (W_HALVES*2)              // 88064
#define BIAS_FLOATS 528
#define WAVE_BASE_BYTE (BIAS_BYTE + BIAS_FLOATS*4)   // 90176
#define WAVE_HALVES 8192                     // 16 KB per wave scratch
#define SMEM_BYTES (WAVE_BASE_BYTE + WAVES*WAVE_HALVES*2)  // 155712

union FragH { v16h v; uint4 q[2]; };

__device__ __forceinline__ v8f vzero(){
  v8f z;
#pragma unroll
  for (int e = 0; e < 8; ++e) z[e] = 0.0f;
  return z;
}

// A-fragment (16xK, 16-bit) gather from LDS buffer laid out [16][KP] halves.
// ISA layout: lanes 0-15 hold M=0..15; half j -> K = kbase + 8*hi + j (j<8),
// K = kbase + 16 + 8*hi + (j-8) (j>=8).  Two aligned 16B chunks per lane.
__device__ __forceinline__ v16h load_a(const _Float16* buf, int KP, int lo, int hi, int kbase){
  FragH f;
  const _Float16* p = buf + lo*KP + kbase + 8*hi;
  f.q[0] = *(const uint4*)(p);
  f.q[1] = *(const uint4*)(p + 16);
  return f.v;
}

// B-fragment pre-packed at init: fragment = 512 halves, lane gets 32B at lane*32.
__device__ __forceinline__ v16h load_b(const _Float16* wbuf, int frag, int lane){
  FragH f;
  const uint4* p = (const uint4*)(wbuf + frag*512 + lane*16);
  f.q[0] = p[0];
  f.q[1] = p[1];
  return f.v;
}

__device__ __forceinline__ v8f wmma16(v16h a, v16h b, v8f c){
  // v_wmma_f32_16x16x32_f16 : D = A(16x32 f16) x B(32x16 f16) + C(16x16 f32)
  return __builtin_amdgcn_wmma_f32_16x16x32_f16(false, a, false, b, (short)0, c, false, false);
}

// Branch-free sigmoid: v_mul + v_exp_f32 + v_add + v_rcp_f32 (no IEEE divide).
__device__ __forceinline__ float sigm(float x){
  return __builtin_amdgcn_rcpf(1.0f + __expf(-x));
}

// Branch-free tanh. gfx1250 has a native V_TANH_F32 transcendental.
__device__ __forceinline__ float fast_tanh(float x){
#if __has_builtin(__builtin_amdgcn_tanhf)
  return __builtin_amdgcn_tanhf(x);
#else
  float xc = fminf(fmaxf(x, -15.0f), 15.0f);   // keep exp finite, branch-free
  float t  = __expf(2.0f * xc);
  return (t - 1.0f) * __builtin_amdgcn_rcpf(t + 1.0f);
#endif
}

// Pack weight (OUT x IN f32, row-major) into WMMA B-fragment order, f16.
// mode 0: output col N maps directly to row o=N (valid if o < OUTvalid)
// mode 1: small GRU; N padded per-gate to 32: gate=N/32, sub=N%32, o=gate*20+sub
// K beyond IN (and invalid cols) are zero-filled => K/N padding contributes 0.
__device__ void pack_weight(_Float16* dst, const float* src, int IN, int KS, int NT,
                            int mode, int OUTvalid, int tid, int nthreads){
  const int total = KS*NT*512;
  for (int idx = tid; idx < total; idx += nthreads){
    int f  = idx >> 9;
    int e  = idx & 511;
    int lane = e >> 4, j = e & 15;
    int hi = lane >> 4, lo = lane & 15;
    int ks = f / NT, nt = f - ks*NT;
    int K = ks*32 + ((j < 8) ? (8*hi + j) : (16 + 8*hi + (j - 8)));
    int N = nt*16 + lo;
    float v = 0.0f;
    int o; bool ok;
    if (mode == 0){ o = N; ok = (o < OUTvalid); }
    else          { int gate = N >> 5; int sub = N & 31; o = gate*20 + sub; ok = (sub < 20); }
    if (ok && K < IN) v = src[o*IN + K];
    dst[idx] = (_Float16)v;
  }
}

// Generic fused GRU step on a 16-row tile.
//  NT_OUT: output 16-col tiles (2 for PH=20 padded to 32, 4 for H=64)
//  KSX/KSH: K-steps (x32) for input / hidden matmuls
//  r,z accumulate x@Wih^T and h@Whh^T together; n parts kept separate.
template<int NT_OUT, int KSX, int KSH, int XKP, int HKP, int NVALID, bool HAS_COPY, bool HAS_OUT>
__device__ __forceinline__ void gru_step(int lane,
    const _Float16* xbuf, const _Float16* hbuf,
    const _Float16* Wx, const _Float16* Wh,
    const float* br, const float* bz, const float* bin, const float* bhn,
    v8f* hstate, _Float16* hnext, _Float16* hcopy, float* gmem)
{
  const int lo = lane & 15, hi = lane >> 4;
  constexpr int NTW = 3*NT_OUT;
  constexpr int HN  = NT_OUT*16;
#pragma unroll
  for (int jt = 0; jt < NT_OUT; ++jt){
    v8f cr = vzero(), cz = vzero(), cin = vzero(), chn = vzero();
#pragma unroll
    for (int ks = 0; ks < KSX; ++ks){
      v16h a = load_a(xbuf, XKP, lo, hi, ks*32);
      cr  = wmma16(a, load_b(Wx, ks*NTW +            jt, lane), cr);
      cz  = wmma16(a, load_b(Wx, ks*NTW +   NT_OUT + jt, lane), cz);
      cin = wmma16(a, load_b(Wx, ks*NTW + 2*NT_OUT + jt, lane), cin);
    }
#pragma unroll
    for (int ks = 0; ks < KSH; ++ks){
      v16h a = load_a(hbuf, HKP, lo, hi, ks*32);
      cr  = wmma16(a, load_b(Wh, ks*NTW +            jt, lane), cr);
      cz  = wmma16(a, load_b(Wh, ks*NTW +   NT_OUT + jt, lane), cz);
      chn = wmma16(a, load_b(Wh, ks*NTW + 2*NT_OUT + jt, lane), chn);
    }
    const int col = jt*16 + lo;
    const float fbr = br[col], fbz = bz[col], fbi = bin[col], fbh = bhn[col];
    v8f hold = hstate[jt];
    v8f hnew;
#pragma unroll
    for (int e = 0; e < 8; ++e){
      float r = sigm(cr[e] + fbr);
      float z = sigm(cz[e] + fbz);
      float n = fast_tanh(cin[e] + fbi + r*(chn[e] + fbh));
      hnew[e] = n + z*(hold[e] - n);     // (1-z)*n + z*h, one fma fewer
    }
    hstate[jt] = hnew;
    const bool valid = (col < NVALID);
#pragma unroll
    for (int e = 0; e < 8; ++e){
      const int m = e + 8*hi;                       // C layout: VGPR e -> M = e + 8*hi
      _Float16 hv = valid ? (_Float16)hnew[e] : (_Float16)0.0f;
      hnext[m*HN + col] = hv;                       // f16 state for next step's A operand
      if (HAS_COPY){ if (valid) hcopy[m*64 + col] = hv; }
      if (HAS_OUT){  if (valid) gmem[m*64 + col] = hnew[e]; }
    }
  }
}

__global__ __launch_bounds__(128) void rnn_scan_kernel(
    const float* __restrict__ seq,
    const float* __restrict__ p1Wih, const float* __restrict__ p1Whh,
    const float* __restrict__ p1bih, const float* __restrict__ p1bhh,
    const float* __restrict__ p2Wih, const float* __restrict__ p2Whh,
    const float* __restrict__ p2bih, const float* __restrict__ p2bhh,
    const float* __restrict__ gW,    const float* __restrict__ gb,
    const float* __restrict__ rWih,  const float* __restrict__ rWhh,
    const float* __restrict__ rbih,  const float* __restrict__ rbhh,
    float* __restrict__ out)
{
  extern __shared__ char smem[];
  _Float16* wh = (_Float16*)smem;
  float*    bf = (float*)(smem + BIAS_BYTE);

  const int tid  = threadIdx.x;
  const int lane = tid & 31;
  const int wv   = tid >> 5;
  const int lo   = lane & 15;

  // ---------------- init: pack weights into WMMA fragments ----------------
  pack_weight(wh+OFF_WP1X, p1Wih, 49, 2,  6, 1,   0, tid, 128);
  pack_weight(wh+OFF_WP1H, p1Whh, 20, 1,  6, 1,   0, tid, 128);
  pack_weight(wh+OFF_WP2X, p2Wih, 49, 2,  6, 1,   0, tid, 128);
  pack_weight(wh+OFF_WP2H, p2Whh, 20, 1,  6, 1,   0, tid, 128);
  pack_weight(wh+OFF_WG,   gW,    48, 2,  1, 0,  16, tid, 128);
  pack_weight(wh+OFF_WRX,  rWih,  42, 2, 12, 0, 192, tid, 128);
  pack_weight(wh+OFF_WRH,  rWhh,  64, 2, 12, 0, 192, tid, 128);

  for (int c = tid; c < 32; c += 128){
    bool v = c < 20;
    bf[  0+c] = v ? p1bih[c]    + p1bhh[c]    : 0.0f;
    bf[ 32+c] = v ? p1bih[20+c] + p1bhh[20+c] : 0.0f;
    bf[ 64+c] = v ? p1bih[40+c]               : 0.0f;
    bf[ 96+c] = v ? p1bhh[40+c]               : 0.0f;
    bf[128+c] = v ? p2bih[c]    + p2bhh[c]    : 0.0f;
    bf[160+c] = v ? p2bih[20+c] + p2bhh[20+c] : 0.0f;
    bf[192+c] = v ? p2bih[40+c]               : 0.0f;
    bf[224+c] = v ? p2bhh[40+c]               : 0.0f;
  }
  for (int c = tid; c < 64; c += 128){
    bf[256+c] = rbih[c]     + rbhh[c];
    bf[320+c] = rbih[64+c]  + rbhh[64+c];
    bf[384+c] = rbih[128+c];
    bf[448+c] = rbhh[128+c];
  }
  for (int c = tid; c < 16; c += 128) bf[512+c] = gb[c];

  // per-wave scratch (f16): A-buffers + double-buffered recurrent state
  _Float16* ws   = wh + (WAVE_BASE_BYTE/2) + wv*WAVE_HALVES;
  _Float16* xp1h = ws;            // [16][64]  x_p1 (cols 49..63 zero)
  _Float16* xp2h = ws + 1024;     // [16][64]  x_p2
  _Float16* inpG = ws + 2048;     // [16][64]  [h1(20)|h2(20)|x_g(8)|pad]
  _Float16* inpR = ws + 3072;     // [16][64]  [g(16)|x_r(26)|pad]
  _Float16* h1b  = ws + 4096;     // 2 x [16][32]
  _Float16* h2b  = ws + 5120;     // 2 x [16][32]
  _Float16* hhb  = ws + 6144;     // 2 x [16][64]
  for (int i = lane; i < WAVE_HALVES; i += 32) ws[i] = (_Float16)0.0f;

  __syncthreads();   // the only barrier: waves share packed weights

  const int wtile = blockIdx.x * WAVES + wv;   // 0..127
  const int b0    = wtile * 16;

  v8f h1f[2], h2f[2], hf[4];
#pragma unroll
  for (int i = 0; i < 2; ++i){ h1f[i] = vzero(); h2f[i] = vzero(); }
#pragma unroll
  for (int i = 0; i < 4; ++i){ hf[i] = vzero(); }

  // ---------------- main serial scan over T ----------------
  for (int t = 0; t < T_STEPS; ++t){
    const float* xg = seq + ((long long)t*BATCH + b0)*DIN;  // 2112 contiguous f32

    // prefetch next timestep's tile (global_prefetch_b8), one line per lane-slot
    if (t + 1 < T_STEPS){
      const float* nx = xg + (long long)BATCH*DIN;
      for (int i = lane*32; i < 16*DIN; i += 32*32)
        __builtin_prefetch(nx + i, 0, 0);
    }

    // stage x tile: coalesced f32 loads, scatter f16 into A-buffers
    for (int i = lane; i < 16*DIN; i += 32){
      int row = i / DIN;
      int c   = i - row*DIN;
      _Float16 hv = (_Float16)xg[i];
      if      (c <  49) xp1h[row*64 + c]            = hv;
      else if (c <  98) xp2h[row*64 + (c-49)]       = hv;
      else if (c < 106) inpG[row*64 + 40 + (c-98)]  = hv;
      else              inpR[row*64 + 16 + (c-106)] = hv;
    }

    const int cur = t & 1, nxt = cur ^ 1;

    // p1 / p2 GRUs (PH=20 padded to 32 cols, Kx=49->64, Kh=20->32)
    gru_step<2,2,1,64,32,20,true,false>(lane, xp1h, h1b + cur*512,
        wh+OFF_WP1X, wh+OFF_WP1H, bf+0, bf+32, bf+64, bf+96,
        h1f, h1b + nxt*512, inpG, nullptr);
    gru_step<2,2,1,64,32,20,true,false>(lane, xp2h, h2b + cur*512,
        wh+OFF_WP2X, wh+OFF_WP2H, bf+128, bf+160, bf+192, bf+224,
        h2f, h2b + nxt*512, inpG + 20, nullptr);

    // gate: g = relu([h1|h2|x_g] @ gW^T + gb), one 16x16 tile, K=48->64
    {
      v8f cg = vzero();
#pragma unroll
      for (int ks = 0; ks < 2; ++ks){
        v16h a = load_a(inpG, 64, lo, lane >> 4, ks*32);
        cg = wmma16(a, load_b(wh+OFF_WG, ks, lane), cg);
      }
      const float gbv = bf[512 + lo];
      const int hi = lane >> 4;
#pragma unroll
      for (int e = 0; e < 8; ++e){
        float g = cg[e] + gbv;
        g = fmaxf(g, 0.0f);
        inpR[(e + 8*hi)*64 + lo] = (_Float16)g;
      }
    }

    // main GRU (H=64, Kx=42->64, Kh=64) + stream f32 hidden to global out
    float* op = out + ((long long)t*BATCH + b0)*64;
    gru_step<4,2,2,64,64,64,false,true>(lane, inpR, hhb + cur*1024,
        wh+OFF_WRX, wh+OFF_WRH, bf+256, bf+320, bf+384, bf+448,
        hf, hhb + nxt*1024, nullptr, op);
  }
}

extern "C" void kernel_launch(void* const* d_in, const int* in_sizes, int n_in,
                              void* d_out, int out_size, void* d_ws, size_t ws_size,
                              hipStream_t stream) {
  (void)in_sizes; (void)n_in; (void)d_ws; (void)ws_size; (void)out_size;
  const float* seq   = (const float*)d_in[0];
  const float* p1Wih = (const float*)d_in[1];
  const float* p1Whh = (const float*)d_in[2];
  const float* p1bih = (const float*)d_in[3];
  const float* p1bhh = (const float*)d_in[4];
  const float* p2Wih = (const float*)d_in[5];
  const float* p2Whh = (const float*)d_in[6];
  const float* p2bih = (const float*)d_in[7];
  const float* p2bhh = (const float*)d_in[8];
  const float* gW    = (const float*)d_in[9];
  const float* gb    = (const float*)d_in[10];
  const float* rWih  = (const float*)d_in[11];
  const float* rWhh  = (const float*)d_in[12];
  const float* rbih  = (const float*)d_in[13];
  const float* rbhh  = (const float*)d_in[14];
  float* out = (float*)d_out;

  (void)hipFuncSetAttribute((const void*)rnn_scan_kernel,
                            hipFuncAttributeMaxDynamicSharedMemorySize, SMEM_BYTES);
  rnn_scan_kernel<<<dim3(BATCH/(16*WAVES)), dim3(32*WAVES), SMEM_BYTES, stream>>>(
      seq, p1Wih, p1Whh, p1bih, p1bhh, p2Wih, p2Whh, p2bih, p2bhh,
      gW, gb, rWih, rWhh, rbih, rbhh, out);
}
